// FastSpatialWindowAttention_56607668962023
// MI455X (gfx1250) — compile-verified, matrix operand order checked by reading, not run on hardware
//
#include <hip/hip_runtime.h>

typedef __attribute__((ext_vector_type(16))) __bf16 v16bf;
typedef __attribute__((ext_vector_type(4)))  __bf16 v4bf;
typedef __attribute__((ext_vector_type(8)))  float  v8f;
typedef __attribute__((ext_vector_type(4)))  int    v4i;

namespace {
constexpr int kH    = 256;
constexpr int kW    = 256;
constexpr int kC    = 256;      // channels
constexpr int kNH   = 8;        // heads
constexpr int kHD   = 32;       // head dim
constexpr int kWS   = 8;        // window side
constexpr int kTok  = 64;       // tokens per window (8*8)
constexpr int kNWin = 4 * 32 * 32; // B * (H/WS) * (W/WS)
constexpr float kScale = 0.17677669529663687f; // 32^-0.5
}

union Frag16 { v16bf v; v4i i[2]; };

// Load a 16x32 bf16 WMMA fragment for this lane.
// Caller passes p already offset to: row(lane)*stride + k0 + (laneHi ? 8 : 0).
// Chunk0 = 8 bf16 at p, chunk1 = 8 bf16 at p+16 (K+16). Both 16B aligned.
__device__ __forceinline__ v16bf load_frag(const __bf16* p) {
  Frag16 f;
  f.i[0] = *reinterpret_cast<const v4i*>(p);
  f.i[1] = *reinterpret_cast<const v4i*>(p + 16);
  return f.v;
}

__device__ __forceinline__ v8f wmma_bf16(v16bf a, v16bf b, v8f c) {
  // D = A(16x32) * B(32x16) + C, fp32 accumulate
  return __builtin_amdgcn_wmma_f32_16x16x32_bf16(false, a, false, b,
                                                 (short)0, c, false, false);
}

// ---------------------------------------------------------------------------
// Kernel 0: fp32 -> bf16 weight conversion into workspace (weights are reused
// by 4096 blocks; keep them bf16 + L2 resident).
// ---------------------------------------------------------------------------
__global__ void convert_weights_kernel(const float* __restrict__ qkv_w,
                                       const float* __restrict__ proj_w,
                                       __bf16* __restrict__ qkv_wb,
                                       __bf16* __restrict__ proj_wb,
                                       int n_qkv, int n_proj) {
  int i = blockIdx.x * blockDim.x + threadIdx.x;
  if (i < n_qkv) qkv_wb[i] = (__bf16)qkv_w[i];
  if (i < n_proj) proj_wb[i] = (__bf16)proj_w[i];
}

// ---------------------------------------------------------------------------
// Fused window attention: one 256-thread block (8 waves) per 8x8 window.
// ---------------------------------------------------------------------------
__global__ __launch_bounds__(256)
void fused_window_attention(const float* __restrict__ x,
                            const __bf16* __restrict__ qkv_wb,
                            const __bf16* __restrict__ proj_wb,
                            const float* __restrict__ proj_b,
                            const float* __restrict__ bias,
                            float* __restrict__ out) {
  __shared__ __bf16 s_q[kTok * kC];          // 32 KB  [token][c] (pre-scaled)
  __shared__ __bf16 s_k[kTok * kC];          // 32 KB  [token][c]
  __shared__ __bf16 s_vT[kC * kTok];         // 32 KB  [c][token] (transposed)
  __shared__ float  s_S[kNH][kTok * kTok];   // 128 KB per-head scores
  __shared__ __bf16 s_P[kNH][kTok * kTok];   // 64 KB  per-head probs (bf16)
  __shared__ float  s_inv[kNH][kTok];        // 2 KB   per-row 1/sum

  // s_x (phase 1-2) and s_O (phase 3-4) overlay the (then-dead) S region.
  __bf16* s_x = reinterpret_cast<__bf16*>(&s_S[0][0]);   // 32 KB
  __bf16* s_O = reinterpret_cast<__bf16*>(&s_S[0][0]);   // 32 KB

  const int tid   = threadIdx.x;
  const int wave  = tid >> 5;
  const int lane  = tid & 31;
  const int lrow  = lane & 15;   // fragment row / N index
  const int lhalf = lane >> 4;   // K-chunk select

  const int wid = blockIdx.x;
  const int b   = wid >> 10;
  const int whh = (wid >> 5) & 31;
  const int www = wid & 31;

  // byte/elem base of window's top-left token in (B,H,W,C)
  const long long rowbase =
      ((long long)(b * kH + whh * kWS) * kW + www * kWS) * kC;

  // ---------------- Phase 1: stage x window into LDS as bf16 ----------------
  {
    const int token = tid >> 2;          // 64 tokens / 4 threads each
    const int part  = (tid & 3) * 64;    // 64 channels per thread
    const int i = token >> 3, j = token & 7;
    const float* src = x + rowbase + ((long long)i * kW + j) * kC + part;
    __bf16* dst = s_x + token * kC + part;
#pragma unroll
    for (int u = 0; u < 64; u += 4) {
      float4 f = *reinterpret_cast<const float4*>(src + u);
      v4bf h;
      h[0] = (__bf16)f.x; h[1] = (__bf16)f.y;
      h[2] = (__bf16)f.z; h[3] = (__bf16)f.w;
      *reinterpret_cast<v4bf*>(dst + u) = h;
    }
  }
  __syncthreads();

  // ---------------- Phase 2: QKV GEMM (64 x 768 x 256) ---------------------
  // 48 N-tiles split round-robin over 8 waves (6 each).
  for (int nt = wave; nt < 48; nt += 8) {
    v8f acc[4] = {};
    const __bf16* wrow = qkv_wb + (nt * 16 + lrow) * kC + lhalf * 8;
    for (int ks = 0; ks < 8; ++ks) {
      v16bf bf = load_frag(wrow + ks * 32);
#pragma unroll
      for (int mt = 0; mt < 4; ++mt) {
        v16bf af = load_frag(s_x + (mt * 16 + lrow) * kC + ks * 32 + lhalf * 8);
        acc[mt] = wmma_bf16(af, bf, acc[mt]);
      }
    }
    // scatter D tiles into q / k / vT (d range uniform per nt)
#pragma unroll
    for (int mt = 0; mt < 4; ++mt) {
#pragma unroll
      for (int r = 0; r < 8; ++r) {
        const int token = mt * 16 + r + lhalf * 8;
        const int d = nt * 16 + lrow;
        const float v = acc[mt][r];
        if (d < 256)       s_q[token * kC + d]            = (__bf16)(v * kScale);
        else if (d < 512)  s_k[token * kC + (d - 256)]    = (__bf16)v;
        else               s_vT[(d - 512) * kTok + token] = (__bf16)v;
      }
    }
  }
  __syncthreads();

  // ---------------- Phase 3: attention, one wave per head ------------------
  const int h = wave;
  float* S = s_S[h];

  // S = q*scale @ k^T + bias   (K = hd = 32 -> single WMMA per 16x16 tile)
  for (int qt = 0; qt < 4; ++qt) {
    v16bf af = load_frag(s_q + (qt * 16 + lrow) * kC + h * kHD + lhalf * 8);
#pragma unroll
    for (int kt = 0; kt < 4; ++kt) {
      v8f acc;
#pragma unroll
      for (int r = 0; r < 8; ++r) {
        const int qtok = qt * 16 + r + lhalf * 8;
        const int ktok = kt * 16 + lrow;
        acc[r] = bias[(h * kTok + qtok) * kTok + ktok];
      }
      v16bf bf = load_frag(s_k + (kt * 16 + lrow) * kC + h * kHD + lhalf * 8);
      acc = wmma_bf16(af, bf, acc);
#pragma unroll
      for (int r = 0; r < 8; ++r)
        S[(qt * 16 + r + lhalf * 8) * kTok + kt * 16 + lrow] = acc[r];
    }
  }
  __syncthreads();

  // softmax: 32 lanes x 2 rows each (unnormalized exp; 1/sum folded into O)
#pragma unroll
  for (int rr = 0; rr < 2; ++rr) {
    const int row = lane + rr * 32;
    float m = -1e30f;
    for (int c = 0; c < kTok; ++c) m = fmaxf(m, S[row * kTok + c]);
    float sum = 0.f;
    for (int c = 0; c < kTok; ++c) {
      float p = __expf(S[row * kTok + c] - m);
      sum += p;
      s_P[h][row * kTok + c] = (__bf16)p;
    }
    s_inv[h][row] = 1.0f / sum;
  }
  __syncthreads();

  // O = P @ v : M=64, N=hd=32 (2 tiles), K=64 (2 steps)
  v8f oacc[4][2] = {};
  for (int ks = 0; ks < 2; ++ks) {
#pragma unroll
    for (int n2 = 0; n2 < 2; ++n2) {
      v16bf bf = load_frag(s_vT + (h * kHD + n2 * 16 + lrow) * kTok +
                           ks * 32 + lhalf * 8);
#pragma unroll
      for (int mt = 0; mt < 4; ++mt) {
        v16bf af = load_frag(&s_P[h][(mt * 16 + lrow) * kTok + ks * 32 + lhalf * 8]);
        oacc[mt][n2] = wmma_bf16(af, bf, oacc[mt][n2]);
      }
    }
  }
  __syncthreads();   // all P/S reads done before s_O overlays the S region

#pragma unroll
  for (int mt = 0; mt < 4; ++mt)
#pragma unroll
    for (int n2 = 0; n2 < 2; ++n2)
#pragma unroll
      for (int r = 0; r < 8; ++r) {
        const int token = mt * 16 + r + lhalf * 8;
        const int dcol  = h * kHD + n2 * 16 + lrow;
        s_O[token * kC + dcol] = (__bf16)(oacc[mt][n2][r] * s_inv[h][token]);
      }
  __syncthreads();

  // ---------------- Phase 4: output projection (64 x 256 x 256) ------------
  for (int nt = wave; nt < 16; nt += 8) {
    v8f acc[4] = {};
    const __bf16* wrow = proj_wb + (nt * 16 + lrow) * kC + lhalf * 8;
    for (int ks = 0; ks < 8; ++ks) {
      v16bf bf = load_frag(wrow + ks * 32);
#pragma unroll
      for (int mt = 0; mt < 4; ++mt) {
        v16bf af = load_frag(s_O + (mt * 16 + lrow) * kC + ks * 32 + lhalf * 8);
        acc[mt] = wmma_bf16(af, bf, acc[mt]);
      }
    }
    const int d = nt * 16 + lrow;
    const float pb = proj_b[d];
#pragma unroll
    for (int mt = 0; mt < 4; ++mt) {
#pragma unroll
      for (int r = 0; r < 8; ++r) {
        const int token = mt * 16 + r + lhalf * 8;
        const int i = token >> 3, j = token & 7;
        out[rowbase + ((long long)i * kW + j) * kC + d] = acc[mt][r] + pb;
      }
    }
  }
}

// ---------------------------------------------------------------------------
extern "C" void kernel_launch(void* const* d_in, const int* in_sizes, int n_in,
                              void* d_out, int out_size, void* d_ws, size_t ws_size,
                              hipStream_t stream) {
  (void)in_sizes; (void)n_in; (void)out_size; (void)ws_size;
  const float* xp     = (const float*)d_in[0];
  const float* qkv_w  = (const float*)d_in[1];
  const float* proj_w = (const float*)d_in[2];
  const float* proj_b = (const float*)d_in[3];
  const float* bias   = (const float*)d_in[4];

  __bf16* qkv_wb  = (__bf16*)d_ws;                                   // 384 KB
  __bf16* proj_wb = (__bf16*)((char*)d_ws + (size_t)3 * kC * kC * 2); // 128 KB

  convert_weights_kernel<<<(3 * kC * kC + 255) / 256, 256, 0, stream>>>(
      qkv_w, proj_w, qkv_wb, proj_wb, 3 * kC * kC, kC * kC);

  fused_window_attention<<<kNWin, 256, 0, stream>>>(
      xp, qkv_wb, proj_wb, proj_b, bias, (float*)d_out);
}
